// DeepSets_34754875359298
// MI455X (gfx1250) — matrix operand order repaired and checked by reading, not compiled, not run on hardware
//
#include <hip/hip_runtime.h>
#include <math.h>
#include <stdint.h>

typedef _Float16 h16;
typedef __attribute__((ext_vector_type(16))) _Float16 v16h;
typedef __attribute__((ext_vector_type(8)))  _Float16 v8h;
typedef __attribute__((ext_vector_type(8)))  float    v8f;

typedef unsigned int u32x4 __attribute__((ext_vector_type(4)));
typedef int          i32x4 __attribute__((ext_vector_type(4)));
typedef int          i32x8 __attribute__((ext_vector_type(8)));

#define LDS_STRIDE 136   // halves per row (272B) -> conflict-free ds_load_b128

#if defined(__HIP_DEVICE_COMPILE__) && __has_builtin(__builtin_amdgcn_tensor_load_to_lds) && \
    __has_builtin(__builtin_amdgcn_s_wait_tensorcnt)
#define HAVE_TDM 1
#else
#define HAVE_TDM 0
#endif

union U16 { v16h v; v8h h[2]; };

__device__ __forceinline__ float gelu_exact(float v) {
    return 0.5f * v * (1.0f + erff(v * 0.70710678118654752f));
}

// low 32 bits of a flat pointer into the LDS aperture = LDS byte offset
__device__ __forceinline__ unsigned lds_byte_offset(const void* p) {
    return (unsigned)(uintptr_t)p;
}

// ---- TDM: DMA `rows` x `halvesPerRow` f16 (contiguous in global) into LDS with
// hardware row padding up to LDS_STRIDE halves. 1-D tile + pad_interval trick. ----
__device__ __forceinline__ void tdm_load_weight(const h16* src, h16* dstLds,
                                                int halvesPerRow, int rows) {
#if HAVE_TDM
    unsigned long long ga = (unsigned long long)(uintptr_t)src;
    unsigned lds = lds_byte_offset(dstLds);
    unsigned total = (unsigned)(halvesPerRow * rows);        // halves (fits 16b tile_dim0)
    int rowDwords = halvesPerRow >> 1;
    unsigned padIntervalCode = (unsigned)(31 - __builtin_clz((unsigned)rowDwords)) - 1u; // 2^(v+1) DW
    unsigned padDwords = (unsigned)((LDS_STRIDE - halvesPerRow) >> 1);
    unsigned padAmountCode = padDwords - 1u;                 // v -> v+1 DWORDs

    u32x4 g0;
    g0[0] = 1u;                                              // count=1 (valid), user mode
    g0[1] = lds;                                             // lds_addr
    g0[2] = (unsigned)(ga & 0xffffffffu);                    // global_addr[31:0]
    g0[3] = (unsigned)((ga >> 32) & 0x01ffffffu) | (2u << 30); // global_addr[56:32] | type=2

    i32x8 g1;
    g1[0] = (int)((1u << 16)                                 // data_size = 2 bytes
                  | (1u << 20)                               // pad_enable
                  | (padIntervalCode << 22)
                  | (padAmountCode << 25));                  // workgroup_mask = 0
    g1[1] = (int)((total & 0xffffu) << 16);                  // tensor_dim0[15:0]
    g1[2] = (int)(((total >> 16) & 0xffffu) | (1u << 16));   // tensor_dim0[31:16] | tensor_dim1=1
    g1[3] = (int)((total & 0xffffu) << 16);                  // tile_dim0
    g1[4] = 0;                                               // tile_dim1=0, tile_dim2=0 (1-D)
    g1[5] = (int)total;                                      // tensor_dim0_stride[31:0]
    g1[6] = 0;
    g1[7] = 0;

    i32x4 gz = {0, 0, 0, 0};
#if __clang_major__ >= 23
    i32x8 gz8 = {0, 0, 0, 0, 0, 0, 0, 0};
    __builtin_amdgcn_tensor_load_to_lds(g0, g1, gz, gz, gz8, 0);
#else
    __builtin_amdgcn_tensor_load_to_lds(g0, g1, gz, gz, 0);
#endif
#else
    (void)src; (void)dstLds; (void)halvesPerRow; (void)rows;
#endif
}

// fallback: cooperative synchronous copy (same LDS layout)
__device__ __forceinline__ void copy_weight_fallback(const h16* src, h16* dstLds,
                                                     int halvesPerRow, int rows, int tid) {
    int chunksPerRow = halvesPerRow >> 3;
    int chunks = rows * chunksPerRow;
    for (int c = tid; c < chunks; c += 256) {
        int row = c / chunksPerRow;
        int col = (c - row * chunksPerRow) << 3;
        *(v8h*)(dstLds + row * LDS_STRIDE + col) = *(const v8h*)(src + row * halvesPerRow + col);
    }
}

#if HAVE_TDM
#define WEIGHT_LOAD(src, dst, hpr, rows) \
    do { if (wave == 0) tdm_load_weight((src), (dst), (hpr), (rows)); } while (0)
#define WEIGHT_WAIT(n) \
    do { if (wave == 0) __builtin_amdgcn_s_wait_tensorcnt(n); } while (0)
#else
#define WEIGHT_LOAD(src, dst, hpr, rows) copy_weight_fallback((src), (dst), (hpr), (rows), tid)
#define WEIGHT_WAIT(n) do { } while (0)
#endif

// ---- WMMA K-loop over tiles: D[16x16] += A[16x32] x B[32x16] ----
template <int NT, int KT>
__device__ __forceinline__ void wmma_tiles(const h16* sH, const h16* sW,
                                           int aRow, int lr, int hi, v8f* acc) {
#pragma unroll
    for (int kk = 0; kk < KT; ++kk) {
        U16 a;
        // A layout (16-bit 16x32): lane=M (both halves), hi selects K offset +8 / +24
        a.h[0] = *(const v8h*)(sH + aRow * LDS_STRIDE + kk * 32 + hi * 8);
        a.h[1] = *(const v8h*)(sH + aRow * LDS_STRIDE + kk * 32 + 16 + hi * 8);
#pragma unroll
        for (int nt = 0; nt < NT; ++nt) {
            U16 b;
            const h16* bp = sW + (nt * 16 + lr) * LDS_STRIDE + kk * 32 + hi * 8;
            b.h[0] = *(const v8h*)(bp);
            b.h[1] = *(const v8h*)(bp + 16);
            acc[nt] = __builtin_amdgcn_wmma_f32_16x16x32_f16(
                false, a.v, false, b.v, (short)0, acc[nt], false, false);
        }
    }
}

// ---- LayerNorm (over NT*16 features) + exact GELU, store f16 back to sH ----
template <int NT>
__device__ __forceinline__ void ln_gelu_store(v8f* acc, h16* sH, int wave, int lr,
                                              int hi, const float* g, const float* be) {
    const float invD = 1.0f / (float)(NT * 16);
    float gv[NT], bev[NT];
#pragma unroll
    for (int nt = 0; nt < NT; ++nt) { gv[nt] = g[nt * 16 + lr]; bev[nt] = be[nt * 16 + lr]; }
#pragma unroll
    for (int r = 0; r < 8; ++r) {
        float s1 = 0.f, s2 = 0.f;
#pragma unroll
        for (int nt = 0; nt < NT; ++nt) { float v = acc[nt][r]; s1 += v; s2 += v * v; }
#pragma unroll
        for (int m = 1; m < 16; m <<= 1) {   // wave32: masks 1,2,4,8 stay within 16-lane half
            s1 += __shfl_xor(s1, m, 32);
            s2 += __shfl_xor(s2, m, 32);
        }
        float mean = s1 * invD;
        float var  = s2 * invD - mean * mean;
        float rs   = rsqrtf(var + 1e-5f);
        int row = wave * 16 + r + hi * 8;
#pragma unroll
        for (int nt = 0; nt < NT; ++nt) {
            float v = (acc[nt][r] - mean) * rs * gv[nt] + bev[nt];
            v = gelu_exact(v);
            sH[row * LDS_STRIDE + nt * 16 + lr] = (h16)v;
        }
    }
}

// =================== phi: fused 4-layer per-point MLP + pooled scatter ===================
__global__ __launch_bounds__(256) void phi_kernel(
    const float* __restrict__ x, const int* __restrict__ idx,
    const h16* __restrict__ W0t, const h16* __restrict__ W1t,
    const h16* __restrict__ W2t, const h16* __restrict__ W3t,
    const float* __restrict__ b0, const float* __restrict__ g0, const float* __restrict__ be0,
    const float* __restrict__ b1, const float* __restrict__ g1, const float* __restrict__ be1,
    const float* __restrict__ b2, const float* __restrict__ g2, const float* __restrict__ be2,
    const float* __restrict__ b3,
    float* __restrict__ pooled, int Npts) {
    __shared__ __align__(16) h16 sH [128 * LDS_STRIDE];  // activations (wave-private rows)
    __shared__ __align__(16) h16 sWa[128 * LDS_STRIDE];  // weight ping
    __shared__ __align__(16) h16 sWb[128 * LDS_STRIDE];  // weight pong

    const int tid  = threadIdx.x;
    const int lane = tid & 31;
    const int wave = tid >> 5;
    const int lr   = lane & 15;
    const int hi   = lane >> 4;
    const int rowBase = blockIdx.x * 128;

    // kick off DMA of layer-1 weights; overlaps x/W0 staging + layer 0 compute
    WEIGHT_LOAD(W1t, sWb, 128, 128);

    // ---- stage x tile [128 x 16] -> f16 in sH cols 0..15, zeros 16..31 ----
    {
        int r = tid >> 1;
        int cHalf = (tid & 1) * 8;
        int grow = rowBase + r;
        float4 v0 = make_float4(0.f, 0.f, 0.f, 0.f);
        float4 v1 = v0;
        if (grow < Npts) {
            const float4* p = (const float4*)(x + (size_t)grow * 16 + cHalf);
            v0 = p[0];
            v1 = p[1];
        }
        h16* dst = &sH[r * LDS_STRIDE + cHalf];
        dst[0] = (h16)v0.x; dst[1] = (h16)v0.y; dst[2] = (h16)v0.z; dst[3] = (h16)v0.w;
        dst[4] = (h16)v1.x; dst[5] = (h16)v1.y; dst[6] = (h16)v1.z; dst[7] = (h16)v1.w;
        h16* z = &sH[r * LDS_STRIDE + 16 + cHalf];
#pragma unroll
        for (int i = 0; i < 8; ++i) z[i] = (h16)0.f;
    }
    // ---- W0t is [128 x 16]; stage into sWa with zero pad for k=16..31 ----
    if (tid < 128) {
        const v8h* s = (const v8h*)(W0t + tid * 16);
        *(v8h*)&sWa[tid * LDS_STRIDE + 0] = s[0];
        *(v8h*)&sWa[tid * LDS_STRIDE + 8] = s[1];
        h16* z = &sWa[tid * LDS_STRIDE + 16];
#pragma unroll
        for (int i = 0; i < 16; ++i) z[i] = (h16)0.f;
    }
    __syncthreads();

    v8f acc[8];
    const int aRow = wave * 16 + lr;

    // ----- layer 0: 16 -> 128 (K padded to 32), weights in sWa -----
#pragma unroll
    for (int nt = 0; nt < 8; ++nt) {
        float bb = b0[nt * 16 + lr];
        v8f a; for (int i = 0; i < 8; ++i) a[i] = bb;
        acc[nt] = a;
    }
    wmma_tiles<8, 1>(sH, sWa, aRow, lr, hi, acc);
    ln_gelu_store<8>(acc, sH, wave, lr, hi, g0, be0);

    // ----- layer 1: 128 -> 128, weights in sWb; DMA W2 -> sWa meanwhile -----
    __syncthreads();                 // everyone done reading sWa
    WEIGHT_LOAD(W2t, sWa, 128, 128);
    WEIGHT_WAIT(1);                  // W1 resident (TDM completes in order)
    __syncthreads();
#pragma unroll
    for (int nt = 0; nt < 8; ++nt) {
        float bb = b1[nt * 16 + lr];
        v8f a; for (int i = 0; i < 8; ++i) a[i] = bb;
        acc[nt] = a;
    }
    wmma_tiles<8, 4>(sH, sWb, aRow, lr, hi, acc);
    ln_gelu_store<8>(acc, sH, wave, lr, hi, g1, be1);

    // ----- layer 2: 128 -> 128, weights in sWa; DMA W3 -> sWb meanwhile -----
    __syncthreads();
    WEIGHT_LOAD(W3t, sWb, 128, 128);
    WEIGHT_WAIT(1);                  // W2 resident
    __syncthreads();
#pragma unroll
    for (int nt = 0; nt < 8; ++nt) {
        float bb = b2[nt * 16 + lr];
        v8f a; for (int i = 0; i < 8; ++i) a[i] = bb;
        acc[nt] = a;
    }
    wmma_tiles<8, 4>(sH, sWa, aRow, lr, hi, acc);
    ln_gelu_store<8>(acc, sH, wave, lr, hi, g2, be2);

    // ----- layer 3: 128 -> 128, no LN/act, scatter-add into pooled -----
    __syncthreads();
    WEIGHT_WAIT(0);                  // W3 resident
    __syncthreads();
#pragma unroll
    for (int nt = 0; nt < 8; ++nt) {
        float bb = b3[nt * 16 + lr];
        v8f a; for (int i = 0; i < 8; ++i) a[i] = bb;
        acc[nt] = a;
    }
    wmma_tiles<8, 4>(sH, sWb, aRow, lr, hi, acc);

#pragma unroll
    for (int r = 0; r < 8; ++r) {
        int grow = rowBase + wave * 16 + r + hi * 8;
        if (grow < Npts) {
            int e = idx[grow];
            float* pe = pooled + (size_t)e * 128 + lr;
#pragma unroll
            for (int nt = 0; nt < 8; ++nt) atomicAdd(pe + nt * 16, acc[nt][r]);
        }
    }
}

// =================== rho: per-event MLP 128 -> 128 -> 64 -> 10 ===================
__global__ __launch_bounds__(256) void rho_kernel(
    const float* __restrict__ pooled, const float* __restrict__ counts,
    const h16* __restrict__ R0t, const h16* __restrict__ R1t, const h16* __restrict__ R2t,
    const float* __restrict__ rb0, const float* __restrict__ rg0, const float* __restrict__ rbe0,
    const float* __restrict__ rb1, const float* __restrict__ rg1, const float* __restrict__ rbe1,
    const float* __restrict__ rb2, float* __restrict__ out) {
    __shared__ __align__(16) h16 sH [128 * LDS_STRIDE];
    __shared__ __align__(16) h16 sWa[128 * LDS_STRIDE];  // R0
    __shared__ __align__(16) h16 sWb[ 64 * LDS_STRIDE];  // R1
    __shared__ __align__(16) h16 sWc[ 16 * LDS_STRIDE];  // R2 (padded rows)

    const int tid  = threadIdx.x;
    const int lane = tid & 31;
    const int wave = tid >> 5;
    const int lr   = lane & 15;
    const int hi   = lane >> 4;
    const int eBase = blockIdx.x * 128;

    // DMA all three weight matrices up front; drain TENSORcnt as layers consume them
    WEIGHT_LOAD(R0t, sWa, 128, 128);
    WEIGHT_LOAD(R1t, sWb, 128, 64);
    WEIGHT_LOAD(R2t, sWc, 64, 16);

    // ---- stage pooled/sqrt(count) -> f16 in sH ----
    {
        int r = tid >> 1;
        int cHalf = (tid & 1) * 64;
        int e = eBase + r;
        float cnt = counts[e];
        float s = (cnt > 0.f) ? rsqrtf(cnt) : 0.f;
        const float4* src = (const float4*)(pooled + (size_t)e * 128 + cHalf);
        h16* dst = &sH[r * LDS_STRIDE + cHalf];
#pragma unroll
        for (int i = 0; i < 16; ++i) {
            float4 v = src[i];
            dst[i * 4 + 0] = (h16)(v.x * s);
            dst[i * 4 + 1] = (h16)(v.y * s);
            dst[i * 4 + 2] = (h16)(v.z * s);
            dst[i * 4 + 3] = (h16)(v.w * s);
        }
    }
    WEIGHT_WAIT(2);                  // R0 resident
    __syncthreads();

    const int aRow = wave * 16 + lr;

    // ----- R0: 128 -> 128, LN + GELU -----
    {
        v8f acc[8];
#pragma unroll
        for (int nt = 0; nt < 8; ++nt) {
            float bb = rb0[nt * 16 + lr];
            v8f a; for (int i = 0; i < 8; ++i) a[i] = bb;
            acc[nt] = a;
        }
        wmma_tiles<8, 4>(sH, sWa, aRow, lr, hi, acc);
        ln_gelu_store<8>(acc, sH, wave, lr, hi, rg0, rbe0);
    }

    // ----- R1: 128 -> 64, LN + GELU -----
    WEIGHT_WAIT(1);                  // R1 resident
    __syncthreads();
    {
        v8f acc[4];
#pragma unroll
        for (int nt = 0; nt < 4; ++nt) {
            float bb = rb1[nt * 16 + lr];
            v8f a; for (int i = 0; i < 8; ++i) a[i] = bb;
            acc[nt] = a;
        }
        wmma_tiles<4, 4>(sH, sWb, aRow, lr, hi, acc);
        ln_gelu_store<4>(acc, sH, wave, lr, hi, rg1, rbe1);
    }

    // ----- R2: 64 -> 10 (N padded to 16 with zero weight rows) -----
    WEIGHT_WAIT(0);                  // R2 resident
    __syncthreads();
    {
        v8f acc[1];
        float bb = (lr < 10) ? rb2[lr] : 0.f;
        v8f a; for (int i = 0; i < 8; ++i) a[i] = bb;
        acc[0] = a;
        wmma_tiles<1, 2>(sH, sWc, aRow, lr, hi, acc);
        if (lr < 10) {
#pragma unroll
            for (int r = 0; r < 8; ++r) {
                int e = eBase + wave * 16 + r + hi * 8;
                out[e * 10 + lr] = acc[0][r];
            }
        }
    }
}

// =================== helpers: zero, counts, weight conversion ===================
__global__ void zero_kernel(float* __restrict__ p, int n) {
    int i = blockIdx.x * blockDim.x + threadIdx.x;
    if (i < n) p[i] = 0.f;
}

__global__ void count_kernel(const int* __restrict__ idx, float* __restrict__ counts, int n) {
    int i = blockIdx.x * blockDim.x + threadIdx.x;
    if (i < n) atomicAdd(&counts[idx[i]], 1.0f);
}

// transpose + fp32 -> f16 for all weight matrices (Wt[n][k] = W[k][n])
__global__ void convert_weights(
    const float* __restrict__ W0, const float* __restrict__ W1,
    const float* __restrict__ W2, const float* __restrict__ W3,
    const float* __restrict__ R0, const float* __restrict__ R1,
    const float* __restrict__ R2,
    h16* __restrict__ W0t, h16* __restrict__ W1t, h16* __restrict__ W2t,
    h16* __restrict__ W3t, h16* __restrict__ R0t, h16* __restrict__ R1t,
    h16* __restrict__ R2t) {
    int i = blockIdx.x * blockDim.x + threadIdx.x;
    if (i < 2048) {                       // W0 [16,128] -> W0t [128,16]
        int n = i >> 4, k = i & 15;
        W0t[i] = (h16)W0[k * 128 + n];
    } else if (i < 2048 + 16384) {        // W1 [128,128]
        int j = i - 2048; int n = j >> 7, k = j & 127;
        W1t[j] = (h16)W1[k * 128 + n];
    } else if (i < 2048 + 32768) {        // W2
        int j = i - (2048 + 16384); int n = j >> 7, k = j & 127;
        W2t[j] = (h16)W2[k * 128 + n];
    } else if (i < 2048 + 49152) {        // W3
        int j = i - (2048 + 32768); int n = j >> 7, k = j & 127;
        W3t[j] = (h16)W3[k * 128 + n];
    } else if (i < 2048 + 49152 + 16384) {  // R0 [128,128]
        int j = i - (2048 + 49152); int n = j >> 7, k = j & 127;
        R0t[j] = (h16)R0[k * 128 + n];
    } else if (i < 2048 + 49152 + 16384 + 8192) {  // R1 [128,64] -> [64,128]
        int j = i - (2048 + 49152 + 16384); int n = j >> 7, k = j & 127;
        R1t[j] = (h16)R1[k * 64 + n];
    } else if (i < 2048 + 49152 + 16384 + 8192 + 1024) {  // R2 [64,10] -> [16,64] zero-pad
        int j = i - (2048 + 49152 + 16384 + 8192); int n = j >> 6, k = j & 63;
        R2t[j] = (n < 10) ? (h16)R2[k * 10 + n] : (h16)0.f;
    }
}

extern "C" void kernel_launch(void* const* d_in, const int* in_sizes, int n_in,
                              void* d_out, int out_size, void* d_ws, size_t ws_size,
                              hipStream_t stream) {
    const float* x   = (const float*)d_in[0];
    const int*   idx = (const int*)d_in[1];
    const float* W0  = (const float*)d_in[2];
    const float* b0  = (const float*)d_in[3];
    const float* g0  = (const float*)d_in[4];
    const float* be0 = (const float*)d_in[5];
    const float* W1  = (const float*)d_in[6];
    const float* b1  = (const float*)d_in[7];
    const float* g1  = (const float*)d_in[8];
    const float* be1 = (const float*)d_in[9];
    const float* W2  = (const float*)d_in[10];
    const float* b2  = (const float*)d_in[11];
    const float* g2  = (const float*)d_in[12];
    const float* be2 = (const float*)d_in[13];
    const float* W3  = (const float*)d_in[14];
    const float* b3  = (const float*)d_in[15];
    const float* R0  = (const float*)d_in[16];
    const float* rb0 = (const float*)d_in[17];
    const float* rg0 = (const float*)d_in[18];
    const float* rbe0= (const float*)d_in[19];
    const float* R1  = (const float*)d_in[20];
    const float* rb1 = (const float*)d_in[21];
    const float* rg1 = (const float*)d_in[22];
    const float* rbe1= (const float*)d_in[23];
    const float* R2  = (const float*)d_in[24];
    const float* rb2 = (const float*)d_in[25];
    float* out = (float*)d_out;

    const int Npts = in_sizes[1];     // idx has one entry per point
    const int E    = out_size / 10;   // 2048

    // workspace layout (~1.21 MB, all 16B-aligned)
    float* pooled = (float*)d_ws;              // E*128 f32
    float* counts = pooled + (size_t)E * 128;  // E f32
    h16* W0t = (h16*)(counts + E);             // 128*16
    h16* W1t = W0t + 2048;                     // 128*128
    h16* W2t = W1t + 16384;
    h16* W3t = W2t + 16384;
    h16* R0t = W3t + 16384;                    // 128*128
    h16* R1t = R0t + 16384;                    // 64*128
    h16* R2t = R1t + 8192;                     // 16*64 (padded)
    (void)ws_size; (void)n_in;

    // 1) zero pooled + counts (contiguous E*129 floats) each launch
    {
        int n = E * 129;
        zero_kernel<<<(n + 255) / 256, 256, 0, stream>>>(pooled, n);
    }
    // 2) convert/transpose weights to f16
    {
        int total = 2048 + 49152 + 16384 + 8192 + 1024;  // 76800
        convert_weights<<<(total + 255) / 256, 256, 0, stream>>>(
            W0, W1, W2, W3, R0, R1, R2, W0t, W1t, W2t, W3t, R0t, R1t, R2t);
    }
    // 3) per-event counts
    count_kernel<<<(Npts + 255) / 256, 256, 0, stream>>>(idx, counts, Npts);
    // 4) fused phi MLP + pooled scatter (128 points / block)
    phi_kernel<<<(Npts + 127) / 128, 256, 0, stream>>>(
        x, idx, W0t, W1t, W2t, W3t,
        b0, g0, be0, b1, g1, be1, b2, g2, be2, b3, pooled, Npts);
    // 5) rho MLP over events (128 events / block)
    rho_kernel<<<E / 128, 256, 0, stream>>>(
        pooled, counts, R0t, R1t, R2t,
        rb0, rg0, rbe0, rb1, rg1, rbe1, rb2, out);
}
